// SSAM_4664334483781
// MI455X (gfx1250) — matrix operand order repaired and checked by reading, not compiled, not run on hardware
//
#include <hip/hip_runtime.h>
#include <hip/hip_bf16.h>

// ---------------------------------------------------------------------------
// Fused self-similarity attention for x:(4,256,64,64) f32.
//   energy = X^T X (per batch, X is C x N, C=256, N=4096)
//   A      = softmax(-energy) row-wise
//   outT   = X @ A^T   (computed directly in C-major output layout)
//   y      = gamma * outT / rowsum + x
//
// Two kernels:
//  1) prep: x(f32) -> ws: Xbf[b][c][n] (bf16) and XqT[b][n][c] = -x (bf16,
//     transposed) so the attention kernel's A-layout Q loads are coalesced.
//  2) flash-attention with online softmax; K/V tiles DMA'd into LDS by the
//     Tensor Data Mover (tensor_load_to_lds, double-buffered, TENSORcnt),
//     GEMMs on v_wmma_f32_16x16x32_bf16.
// ---------------------------------------------------------------------------

typedef __bf16 v16bf __attribute__((ext_vector_type(16)));
typedef __bf16 v8bf  __attribute__((ext_vector_type(8)));
typedef float  v8f   __attribute__((ext_vector_type(8)));
typedef unsigned int u32x4 __attribute__((ext_vector_type(4)));
typedef int          i32x8 __attribute__((ext_vector_type(8)));
typedef int          i32x4 __attribute__((ext_vector_type(4)));

#define BATCH 4
#define CDIM  256
#define NDIM  4096   // 64*64
#define MTILE 64     // m-columns staged per iteration
#define NWAVE 4      // waves per block, each owns 16 query rows

// ---- prep: f32 -> bf16 copies (normal + negated-transposed) ----------------
__global__ __launch_bounds__(256)
void ssam_prep_bf16(const float* __restrict__ x,
                    __bf16* __restrict__ xbf,
                    __bf16* __restrict__ xqT) {
    const size_t total = (size_t)BATCH * CDIM * NDIM;
    for (size_t i = (size_t)blockIdx.x * blockDim.x + threadIdx.x;
         i < total; i += (size_t)gridDim.x * blockDim.x) {
        const float v = x[i];
        xbf[i] = (__bf16)v;
        const size_t b = i / ((size_t)CDIM * NDIM);
        const size_t r = i - b * ((size_t)CDIM * NDIM);
        const size_t c = r / NDIM;
        const size_t n = r - c * NDIM;
        xqT[(b * NDIM + n) * CDIM + c] = (__bf16)(-v);
    }
}

// ---- TDM: DMA one 256x64 bf16 tile (row stride NDIM) into LDS --------------
// D# bit layout per CDNA5 ISA ch.8: group0 = {count,lds_addr,global_addr,type},
// group1 = {data_size, tensor_dim0/1, tile_dim0/1/2, tensor_dim0_stride}.
// This toolchain exposes the 6-arg builtin (extra int32x8 group, zero-filled).
__device__ __forceinline__ void tdm_load_tile(unsigned lds_addr,
                                              const __bf16* gptr) {
    const unsigned long long ga = (unsigned long long)(uintptr_t)gptr;
    u32x4 g0;
    g0[0] = 1u;                                            // count=1 (valid)
    g0[1] = lds_addr;                                      // LDS byte address
    g0[2] = (unsigned)ga;                                  // global_addr[31:0]
    g0[3] = ((unsigned)(ga >> 32) & 0x01FFFFFFu) | (2u << 30); // [56:32]|type=2
    i32x8 g1;
    g1[0] = (1 << 16);                                     // data_size=1 (2B)
    g1[1] = (int)((NDIM & 0xFFFF) << 16);                  // tensor_dim0 lo16
    g1[2] = (int)(((NDIM >> 16) & 0xFFFF) | ((CDIM & 0xFFFF) << 16)); // dim0 hi / dim1 lo
    g1[3] = (int)(((CDIM >> 16) & 0xFFFF) | (MTILE << 16));// dim1 hi / tile_dim0
    g1[4] = (int)(CDIM & 0xFFFF);                          // tile_dim1=256, tile_dim2=0
    g1[5] = (int)NDIM;                                     // tensor_dim0_stride lo32
    g1[6] = 0;                                             // stride hi / dim1_stride lo
    g1[7] = 0;
    i32x4 g2 = {0, 0, 0, 0};                               // 2-D tile: unused
    i32x4 g3 = {0, 0, 0, 0};
    i32x8 g4 = {0, 0, 0, 0, 0, 0, 0, 0};                   // extra group (zeroed)
    __builtin_amdgcn_tensor_load_to_lds(g0, g1, g2, g3, g4, 0);
}

// ---- fused flash attention --------------------------------------------------
__global__ __launch_bounds__(128)
void ssam_flash_wmma(const float* __restrict__ x,
                     const __bf16* __restrict__ xbf,
                     const __bf16* __restrict__ xqT,
                     const float* __restrict__ gamma,
                     float* __restrict__ y) {
    __shared__ __bf16 lds_k[2][CDIM * MTILE];         // 2 x 32 KB, double-buffered
    __shared__ __bf16 lds_p[NWAVE][MTILE * 16];       // 8 KB
    __shared__ float  lds_stat[NWAVE][16];

    const int tid  = threadIdx.x;
    const int w    = tid >> 5;          // wave id 0..3
    const int lane = tid & 31;
    const int half = lane >> 4;
    const int l16  = lane & 15;

    const int b  = blockIdx.y;
    const int n0 = blockIdx.x * 64;
    const int nw = n0 + w * 16;

    const __bf16* Xbfb = xbf + (size_t)b * CDIM * NDIM;

    // ---- Q strip in WMMA A-layout, from negated-transposed bf16 copy -------
    // lane L<16 -> M=L, K in {0..7,16..23}; lane>=16 -> M=L-16, K in {8..15,24..31}
    v16bf qa[8];
    {
        const int Mrow = nw + l16;
        const __bf16* qrow = xqT + ((size_t)b * NDIM + Mrow) * CDIM;
        #pragma unroll
        for (int kk = 0; kk < 8; ++kk) {
            const v8bf lo = *(const v8bf*)&qrow[kk * 32 + half * 8];
            const v8bf hi = *(const v8bf*)&qrow[kk * 32 + 16 + half * 8];
            #pragma unroll
            for (int i = 0; i < 8; ++i) { qa[kk][i] = lo[i]; qa[kk][8 + i] = hi[i]; }
        }
    }

    // ---- Accumulators -------------------------------------------------------
    v8f O[16];
    #pragma unroll
    for (int ct = 0; ct < 16; ++ct) O[ct] = {};
    float mrun[8], lrun[8];
    #pragma unroll
    for (int v = 0; v < 8; ++v) { mrun[v] = -1e30f; lrun[v] = 0.0f; }

    // ---- Prologue: TDM-load first K/V tile (wave 0 only; TDM ignores EXEC) --
    if (w == 0) {
        tdm_load_tile((unsigned)(uintptr_t)&lds_k[0][0], Xbfb /* m0 = 0 */);
    }

    for (int mb = 0; mb < NDIM / MTILE; ++mb) {
        const int cur = mb & 1;
        const __bf16* kt = &lds_k[cur][0];

        if (w == 0) __builtin_amdgcn_s_wait_tensorcnt(0);  // tile `cur` landed
        __syncthreads();                                   // publish to all waves

        // Prefetch next tile into the other buffer while we compute.
        if (w == 0 && mb + 1 < NDIM / MTILE) {
            tdm_load_tile((unsigned)(uintptr_t)&lds_k[cur ^ 1][0],
                          Xbfb + (size_t)(mb + 1) * MTILE);
        }

        // GEMM1: S[16n x 64m] = (-Q) . K   (K-dim C = 256)
        v8f S[4];
        #pragma unroll
        for (int mt = 0; mt < 4; ++mt) {
            v8f s = {};
            #pragma unroll
            for (int kk = 0; kk < 8; ++kk) {
                // B 32x16 layout: lane -> N=m (mt*16+l16), K rows {8h..8h+7, 16+8h..16+8h+7}
                v16bf bm;
                #pragma unroll
                for (int i = 0; i < 8; ++i) {
                    bm[i]     = kt[(kk * 32 + half * 8 + i) * MTILE + mt * 16 + l16];
                    bm[8 + i] = kt[(kk * 32 + 16 + half * 8 + i) * MTILE + mt * 16 + l16];
                }
                s = __builtin_amdgcn_wmma_f32_16x16x32_bf16(
                        false, qa[kk], false, bm, (short)0, s, false, false);
            }
            S[mt] = s;
        }

        // ---- Online softmax (component v <-> row v + 8*half) ---------------
        v8f cmax = S[0];
        #pragma unroll
        for (int mt = 1; mt < 4; ++mt)
            #pragma unroll
            for (int v = 0; v < 8; ++v) cmax[v] = fmaxf(cmax[v], S[mt][v]);
        #pragma unroll
        for (int d = 1; d < 16; d <<= 1)
            #pragma unroll
            for (int v = 0; v < 8; ++v)
                cmax[v] = fmaxf(cmax[v], __shfl_xor(cmax[v], d, 32));

        float mnew[8], corr_c[8];
        #pragma unroll
        for (int v = 0; v < 8; ++v) {
            mnew[v]   = fmaxf(mrun[v], cmax[v]);
            corr_c[v] = __expf(mrun[v] - mnew[v]);
            mrun[v]   = mnew[v];
        }

        v8f rsum = {};
        #pragma unroll
        for (int mt = 0; mt < 4; ++mt)
            #pragma unroll
            for (int v = 0; v < 8; ++v) {
                const float p = __expf(S[mt][v] - mnew[v]);
                S[mt][v] = p;
                rsum[v] += p;
            }
        #pragma unroll
        for (int d = 1; d < 16; d <<= 1)
            #pragma unroll
            for (int v = 0; v < 8; ++v) rsum[v] += __shfl_xor(rsum[v], d, 32);
        #pragma unroll
        for (int v = 0; v < 8; ++v) lrun[v] = lrun[v] * corr_c[v] + rsum[v];

        // Re-index corr from component(row) form to per-lane(row = l16) form.
        if (l16 == 0) {
            #pragma unroll
            for (int v = 0; v < 8; ++v) lds_stat[w][half * 8 + v] = corr_c[v];
        }
        asm volatile("s_wait_dscnt 0" ::: "memory");   // wave-local LDS RAW
        const float corr_lane = lds_stat[w][l16];
        #pragma unroll
        for (int ct = 0; ct < 16; ++ct)
            #pragma unroll
            for (int v = 0; v < 8; ++v) O[ct][v] *= corr_lane;

        // P -> per-wave LDS in [m][n] order (bf16) for B-layout reads.
        #pragma unroll
        for (int mt = 0; mt < 4; ++mt) {
            __bf16 tmp[8];
            #pragma unroll
            for (int v = 0; v < 8; ++v) tmp[v] = (__bf16)S[mt][v];
            *(v8bf*)&lds_p[w][(mt * 16 + l16) * 16 + half * 8] = *(v8bf*)tmp;
        }
        asm volatile("s_wait_dscnt 0" ::: "memory");

        // GEMM2: outT[c, n] += X[c, m] * P[n, m]   (K-dim = 64 m)
        #pragma unroll
        for (int mm = 0; mm < 2; ++mm) {
            v16bf pb;
            #pragma unroll
            for (int i = 0; i < 8; ++i) {
                pb[i]     = lds_p[w][(mm * 32 + half * 8 + i) * 16 + l16];
                pb[8 + i] = lds_p[w][(mm * 32 + 16 + half * 8 + i) * 16 + l16];
            }
            #pragma unroll
            for (int ct = 0; ct < 16; ++ct) {
                const int base = (ct * 16 + l16) * MTILE + mm * 32 + (half ? 8 : 0);
                const v8bf lo = *(const v8bf*)&kt[base];
                const v8bf hi = *(const v8bf*)&kt[base + 16];
                v16bf am;
                #pragma unroll
                for (int i = 0; i < 8; ++i) { am[i] = lo[i]; am[8 + i] = hi[i]; }
                O[ct] = __builtin_amdgcn_wmma_f32_16x16x32_bf16(
                            false, am, false, pb, (short)0, O[ct], false, false);
            }
        }
        __syncthreads();   // done reading `cur` before it becomes the DMA target
    }

    // ---- Epilogue: divide by row sum, scale by gamma, add residual ---------
    if (l16 == 0) {
        #pragma unroll
        for (int v = 0; v < 8; ++v) lds_stat[w][half * 8 + v] = lrun[v];
    }
    asm volatile("s_wait_dscnt 0" ::: "memory");
    const float linv = 1.0f / lds_stat[w][l16];
    const float g = gamma[0];
    const int n = nw + l16;
    #pragma unroll
    for (int ct = 0; ct < 16; ++ct) {
        #pragma unroll
        for (int v = 0; v < 8; ++v) {
            const int c = ct * 16 + v + 8 * half;
            const size_t idx = (size_t)b * CDIM * NDIM + (size_t)c * NDIM + n;
            y[idx] = g * (O[ct][v] * linv) + x[idx];
        }
    }
}

extern "C" void kernel_launch(void* const* d_in, const int* in_sizes, int n_in,
                              void* d_out, int out_size, void* d_ws, size_t ws_size,
                              hipStream_t stream) {
    const float* x     = (const float*)d_in[0];
    const float* gamma = (const float*)d_in[1];
    float* y           = (float*)d_out;
    (void)in_sizes; (void)n_in; (void)out_size; (void)ws_size;

    // ws layout: Xbf (8 MB) | XqT (8 MB)
    __bf16* xbf = (__bf16*)d_ws;
    __bf16* xqT = xbf + (size_t)BATCH * CDIM * NDIM;

    ssam_prep_bf16<<<dim3(2048), dim3(256), 0, stream>>>(x, xbf, xqT);

    dim3 grid(NDIM / 64, BATCH);
    ssam_flash_wmma<<<grid, dim3(128), 0, stream>>>(x, xbf, xqT, gamma, y);
}
